// SelfAttention_10565619548494
// MI455X (gfx1250) — matrix operand order repaired and checked by reading, not compiled, not run on hardware
//
#include <hip/hip_runtime.h>
#include <hip/hip_bf16.h>

// ---------------------------------------------------------------------------
// Types for WMMA fragments (gfx1250, wave32)
// ---------------------------------------------------------------------------
typedef _Float16 h4  __attribute__((ext_vector_type(4)));
typedef _Float16 h8  __attribute__((ext_vector_type(8)));
typedef _Float16 h16 __attribute__((ext_vector_type(16)));
typedef float    f8  __attribute__((ext_vector_type(8)));

#define B_  4
#define S_  1024
#define D_  1024
#define H_  16
#define DH_ 64

__device__ __forceinline__ h16 cat8(h8 lo, h8 hi) {
  return __builtin_shufflevector(lo, hi, 0,1,2,3,4,5,6,7,8,9,10,11,12,13,14,15);
}

// A fragment (16x32 f16, MxK): lane m = lane&15, half = lane>>4.
// elements 0..7  = row[k0 + 8*half + 0..7]
// elements 8..15 = row[k0 + 16 + 8*half + 0..7]  -> two 16B vector loads
__device__ __forceinline__ h16 ldA(const _Float16* rowk0, int half) {
  const h8* lo = (const h8*)(rowk0 + 8 * half);
  const h8* hi = (const h8*)(rowk0 + 16 + 8 * half);
  return cat8(*lo, *hi);
}

// B fragment (32x16 f16, KxN) from a contiguous 32-element K-run:
// elements 0..7 = run[8*half + 0..7], 8..15 = run[16 + 8*half + 0..7]  -> two 16B vector loads
__device__ __forceinline__ h16 ldBrun(const _Float16* runk0, int half) {
  const h8* lo = (const h8*)(runk0 + 8 * half);
  const h8* hi = (const h8*)(runk0 + 16 + 8 * half);
  return cat8(*lo, *hi);
}

__device__ __forceinline__ f8 wmma16(h16 a, h16 b, f8 c) {
  return __builtin_amdgcn_wmma_f32_16x16x32_f16(false, a, false, b,
                                                (short)0, c, false, false);
}

// ---------------------------------------------------------------------------
// f32 -> f16 conversion (vectorized x4) for X
// ---------------------------------------------------------------------------
__global__ __launch_bounds__(256) void cvt_f32_f16(const float* __restrict__ in,
                                                   _Float16* __restrict__ out,
                                                   int n4) {
  int i = blockIdx.x * blockDim.x + threadIdx.x;
  if (i < n4) {
    float4 v = ((const float4*)in)[i];
    h4 r = { (_Float16)v.x, (_Float16)v.y, (_Float16)v.z, (_Float16)v.w };
    ((h4*)out)[i] = r;
  }
}

// ---------------------------------------------------------------------------
// Weight convert + K-panel repack:  W[k][n] f32  ->  Wp[(k/32)*N + n][k%32] f16
// Makes every WMMA B-fragment a contiguous 32-half run (2x b128 loads).
// ---------------------------------------------------------------------------
__global__ __launch_bounds__(256) void wrepack(const float* __restrict__ in,
                                               _Float16* __restrict__ out) {
  int idx = blockIdx.x * blockDim.x + threadIdx.x;   // 1Mi threads
  int k = idx >> 10, n = idx & (D_ - 1);
  out[((size_t)(k >> 5) * D_ + n) * 32 + (k & 31)] = (_Float16)in[idx];
}

// ---------------------------------------------------------------------------
// GEMM: Xh[4096x1024] f16 @ Wp (k-panel f16) -> f16 out.
// One wave computes 32x64 (2 A frags, 4 shared B frags, 8 accumulators/k-step).
// vt=0: out is [B,H,S,DH] (Q,K);  vt=1: out is [B,H,DH,S] (V transposed).
// ---------------------------------------------------------------------------
__global__ __launch_bounds__(128) void proj_gemm(const _Float16* __restrict__ X,
                                                 const _Float16* __restrict__ Wp,
                                                 _Float16* __restrict__ out,
                                                 int vt) {
  const int K = D_, N = D_;
  int w = (blockIdx.x * blockDim.x + threadIdx.x) >> 5;
  int lane = threadIdx.x & 31;
  int nl = lane & 15, half = lane >> 4;
  int nt = w >> 7;                       // 16 col tiles of 64
  int mt = w & 127;                      // 128 row tiles of 32 (block-adjacent)
  int m0 = mt * 32, n0 = nt * 64;

  f8 acc[2][4] = {};
  const _Float16* xr0 = X + (size_t)(m0 + nl) * K;
  const _Float16* xr1 = X + (size_t)(m0 + 16 + nl) * K;

  for (int k0 = 0; k0 < K; k0 += 32) {
    h16 a0 = ldA(xr0 + k0, half);
    h16 a1 = ldA(xr1 + k0, half);
    const _Float16* wp = Wp + ((size_t)(k0 >> 5) * N + n0 + nl) * 32;
#pragma unroll
    for (int t = 0; t < 4; ++t) {
      h16 b = ldBrun(wp + (size_t)t * 16 * 32, half);
      acc[0][t] = wmma16(a0, b, acc[0][t]);
      acc[1][t] = wmma16(a1, b, acc[1][t]);
    }
  }
#pragma unroll
  for (int r2 = 0; r2 < 2; ++r2) {
#pragma unroll
    for (int t = 0; t < 4; ++t) {
#pragma unroll
      for (int i = 0; i < 8; ++i) {
        int r = m0 + r2 * 16 + i + 8 * half;   // global row: b*S + s
        int n = n0 + t * 16 + nl;              // col: h*64 + dh
        int bb = r >> 10, s = r & 1023, hh = n >> 6, dh = n & 63;
        size_t o = vt ? ((size_t)(bb * H_ + hh) * DH_ + dh) * S_ + s
                      : ((size_t)(bb * H_ + hh) * S_ + s) * DH_ + dh;
        out[o] = (_Float16)acc[r2][t][i];
      }
    }
  }
}

// ---------------------------------------------------------------------------
// Output GEMM: Hh[4096x1024] f16 @ Wop (k-panel f16) -> f32 out [4096x1024]
// ---------------------------------------------------------------------------
__global__ __launch_bounds__(128) void out_gemm(const _Float16* __restrict__ X,
                                                const _Float16* __restrict__ Wp,
                                                float* __restrict__ out) {
  const int K = D_, N = D_;
  int w = (blockIdx.x * blockDim.x + threadIdx.x) >> 5;
  int lane = threadIdx.x & 31;
  int nl = lane & 15, half = lane >> 4;
  int nt = w >> 7;
  int mt = w & 127;
  int m0 = mt * 32, n0 = nt * 64;

  f8 acc[2][4] = {};
  const _Float16* xr0 = X + (size_t)(m0 + nl) * K;
  const _Float16* xr1 = X + (size_t)(m0 + 16 + nl) * K;

  for (int k0 = 0; k0 < K; k0 += 32) {
    h16 a0 = ldA(xr0 + k0, half);
    h16 a1 = ldA(xr1 + k0, half);
    const _Float16* wp = Wp + ((size_t)(k0 >> 5) * N + n0 + nl) * 32;
#pragma unroll
    for (int t = 0; t < 4; ++t) {
      h16 b = ldBrun(wp + (size_t)t * 16 * 32, half);
      acc[0][t] = wmma16(a0, b, acc[0][t]);
      acc[1][t] = wmma16(a1, b, acc[1][t]);
    }
  }
#pragma unroll
  for (int r2 = 0; r2 < 2; ++r2) {
#pragma unroll
    for (int t = 0; t < 4; ++t) {
#pragma unroll
      for (int i = 0; i < 8; ++i) {
        int r = m0 + r2 * 16 + i + 8 * half;
        int n = n0 + t * 16 + nl;
        out[(size_t)r * N + n] = acc[r2][t][i];
      }
    }
  }
}

// ---------------------------------------------------------------------------
// Flash attention with additive interaction bias + key mask.
// Q,K in [B,H,S,64]; V transposed in [B,H,64,S] so PV B-frags are contiguous.
// Per 32-key step: 4 WMMA scores + 4 WMMA PV; bias stream is non-temporal and
// prefetched 2 tiles ahead (global_prefetch_b8).
// ---------------------------------------------------------------------------
__global__ __launch_bounds__(128) void attn_kernel(
    const _Float16* __restrict__ Q, const _Float16* __restrict__ Km,
    const _Float16* __restrict__ Vt, const float* __restrict__ inter,
    const unsigned char* __restrict__ mask, _Float16* __restrict__ hid) {
  __shared__ _Float16 ldsP[4][16 * 32];

  int warp = threadIdx.x >> 5, lane = threadIdx.x & 31;
  int nl = lane & 15, half = lane >> 4;
  int wave = blockIdx.x * 4 + warp;     // 0..4095
  int qt = wave & 63;                   // 64 q-tiles per (b,h)
  int bh = wave >> 6;
  int b = bh >> 4, h = bh & 15;
  int q0 = qt * 16;

  const _Float16* Qb = Q + (size_t)bh * S_ * DH_;
  const _Float16* Kb = Km + (size_t)bh * S_ * DH_;
  const _Float16* Vb = Vt + (size_t)bh * DH_ * S_;   // [64][S]

  const _Float16* qrow = Qb + (size_t)(q0 + nl) * DH_;
  h16 aq0 = ldA(qrow, half);
  h16 aq1 = ldA(qrow + 32, half);

  f8 acc[4] = {};
  float rmax[8], rsum[8];
#pragma unroll
  for (int i = 0; i < 8; ++i) { rmax[i] = -1e30f; rsum[i] = 0.f; }

  _Float16* P = ldsP[warp];
  const float qk_scale = 0.125f;        // 1/sqrt(64)
  const float* interb = inter + ((size_t)(b * S_ + q0) * S_) * H_ + h;

  for (int kt = 0; kt < S_ / 32; ++kt) {
    int key0 = kt * 32;
    // prefetch the bias stream 2 key-tiles ahead (wraps; speculative-safe)
    {
      int kpf = (key0 + 64) & (S_ - 1);
#pragma unroll
      for (int i = 0; i < 8; ++i) {
        const float* pf =
            interb + ((size_t)(i + 8 * half) * S_ + kpf + nl) * H_;
        __builtin_prefetch(pf, 0, 0);
        __builtin_prefetch(pf + 16 * H_, 0, 0);
      }
    }
    f8 s[2] = {};
    // ---- scores: Q @ K^T (contiguous K-runs of row-major K)
#pragma unroll
    for (int c = 0; c < 2; ++c) {
      const _Float16* krow = Kb + (size_t)(key0 + c * 16 + nl) * DH_;
      s[c] = wmma16(aq0, ldBrun(krow, half), s[c]);
      s[c] = wmma16(aq1, ldBrun(krow + 32, half), s[c]);
    }
    // ---- scale + interaction bias (non-temporal) + mask
#pragma unroll
    for (int c = 0; c < 2; ++c) {
      int key = key0 + c * 16 + nl;
      float mterm = mask[b * S_ + key] ? 0.f : -1e9f;
      const float* ib = interb + (size_t)key * H_;
#pragma unroll
      for (int i = 0; i < 8; ++i) {
        float bias =
            __builtin_nontemporal_load(ib + (size_t)(i + 8 * half) * S_ * H_);
        s[c][i] = s[c][i] * qk_scale + bias + mterm;
      }
    }
    // ---- online softmax (row = i + 8*half across each 16-lane half)
#pragma unroll
    for (int i = 0; i < 8; ++i) {
      float mx = fmaxf(s[0][i], s[1][i]);
#pragma unroll
      for (int off = 1; off < 16; off <<= 1)
        mx = fmaxf(mx, __shfl_xor(mx, off, 16));
      float nm = fmaxf(rmax[i], mx);
      float resc = __expf(rmax[i] - nm);
      rmax[i] = nm;
      float p0 = __expf(s[0][i] - nm);
      float p1 = __expf(s[1][i] - nm);
      float ps = p0 + p1;
#pragma unroll
      for (int off = 1; off < 16; off <<= 1) ps += __shfl_xor(ps, off, 16);
      rsum[i] = rsum[i] * resc + ps;
#pragma unroll
      for (int t = 0; t < 4; ++t) acc[t][i] *= resc;
      P[(i + 8 * half) * 32 + nl] = (_Float16)p0;
      P[(i + 8 * half) * 32 + 16 + nl] = (_Float16)p1;
    }
    asm volatile("s_wait_dscnt 0" ::: "memory");
    h16 ap;
    {
      const h8* lo = (const h8*)(P + nl * 32 + 8 * half);
      const h8* hi = (const h8*)(P + nl * 32 + 16 + 8 * half);
      ap = cat8(*lo, *hi);
    }
    // ---- acc += P @ V : contiguous key-runs of V^T rows (2x b128 each)
#pragma unroll
    for (int t = 0; t < 4; ++t) {
      h16 bv = ldBrun(Vb + (size_t)(t * 16 + nl) * S_ + key0, half);
      acc[t] = wmma16(ap, bv, acc[t]);
    }
  }
  // ---- normalize and write hidden [b, s, h*64+dh] as f16
#pragma unroll
  for (int i = 0; i < 8; ++i) {
    float inv = 1.f / rsum[i];
    int q = q0 + i + 8 * half;
#pragma unroll
    for (int t = 0; t < 4; ++t)
      hid[(size_t)(b * S_ + q) * D_ + h * DH_ + t * 16 + nl] =
          (_Float16)(acc[t][i] * inv);
  }
}

// ---------------------------------------------------------------------------
// Host-side launch
// ---------------------------------------------------------------------------
extern "C" void kernel_launch(void* const* d_in, const int* in_sizes, int n_in,
                              void* d_out, int out_size, void* d_ws,
                              size_t ws_size, hipStream_t stream) {
  const float* x = (const float*)d_in[0];
  const unsigned char* mask = (const unsigned char*)d_in[1];
  const float* inter = (const float*)d_in[2];
  const float* wq = (const float*)d_in[3];
  const float* wk = (const float*)d_in[4];
  const float* wv = (const float*)d_in[5];
  const float* wo = (const float*)d_in[6];
  float* out = (float*)d_out;

  const size_t nX = (size_t)B_ * S_ * D_;   // 4 Mi elements
  const size_t nW = (size_t)D_ * D_;        // 1 Mi elements

  char* ws = (char*)d_ws;
  _Float16* Xh  = (_Float16*)ws; ws += nX * sizeof(_Float16);
  _Float16* Qh  = (_Float16*)ws; ws += nX * sizeof(_Float16);
  _Float16* Kh  = (_Float16*)ws; ws += nX * sizeof(_Float16);
  _Float16* Vth = (_Float16*)ws; ws += nX * sizeof(_Float16);
  _Float16* Hh  = (_Float16*)ws; ws += nX * sizeof(_Float16);
  _Float16* Wqp = (_Float16*)ws; ws += nW * sizeof(_Float16);
  _Float16* Wkp = (_Float16*)ws; ws += nW * sizeof(_Float16);
  _Float16* Wvp = (_Float16*)ws; ws += nW * sizeof(_Float16);
  _Float16* Wop = (_Float16*)ws; ws += nW * sizeof(_Float16);

  // X: f32 -> f16; weights: f32 -> f16 + k-panel repack
  cvt_f32_f16<<<(int)(nX / 4 / 256), 256, 0, stream>>>(x, Xh, (int)(nX / 4));
  wrepack<<<(int)(nW / 256), 256, 0, stream>>>(wq, Wqp);
  wrepack<<<(int)(nW / 256), 256, 0, stream>>>(wk, Wkp);
  wrepack<<<(int)(nW / 256), 256, 0, stream>>>(wv, Wvp);
  wrepack<<<(int)(nW / 256), 256, 0, stream>>>(wo, Wop);

  // QKV projections: 2048 wave tiles (32x64) = 512 blocks x 4 waves
  proj_gemm<<<512, 128, 0, stream>>>(Xh, Wqp, Qh, 0);
  proj_gemm<<<512, 128, 0, stream>>>(Xh, Wkp, Kh, 0);
  proj_gemm<<<512, 128, 0, stream>>>(Xh, Wvp, Vth, 1);   // V stored transposed

  // flash attention: 4096 waves (B*H*S/16)
  attn_kernel<<<1024, 128, 0, stream>>>(Qh, Kh, Vth, inter, mask, Hh);

  // output projection to f32
  out_gemm<<<512, 128, 0, stream>>>(Hh, Wop, out);
}
